// BertAttention_12498354832077
// MI455X (gfx1250) — matrix operand order repaired and checked by reading, not compiled, not run on hardware
//
#include <hip/hip_runtime.h>
#include <hip/hip_bf16.h>
#include <stdint.h>

// ---------------------------------------------------------------------------
// BERT self-attention block for gfx1250 (MI455X).
// bf16 WMMA + f32 accumulate; async global->LDS staging (ASYNCcnt) for shared
// GEMM operands; LDS relayout for the P*V^T A-operand.
// ---------------------------------------------------------------------------

#define B_ 4
#define S_ 1024
#define H_ 12
#define DH_ 64
#define E_ 768
#define DM_ 768
#define LN_EPS 1e-12f

typedef __bf16 bf16;
typedef __attribute__((ext_vector_type(16))) __bf16 v16bf;
typedef __attribute__((ext_vector_type(8)))  __bf16 v8bf;
typedef __attribute__((ext_vector_type(8)))  float  v8f;

__device__ __forceinline__ bf16 f2bf(float f) {
  unsigned u = __builtin_bit_cast(unsigned, f);
  unsigned r = u + 0x7fffu + ((u >> 16) & 1u);        // round-to-nearest-even
  return __builtin_bit_cast(bf16, (unsigned short)(r >> 16));
}

// A-operand fragment (16x32 bf16): lanes 0-15 hold contiguous K 0..15,
// lanes 16-31 hold K 16..31 of row M=lane&15. One 32B load per lane.
__device__ __forceinline__ v16bf load_a_frag(const bf16* rowbase, int hi) {
  return *(const v16bf*)(rowbase + hi * 16);
}

// B-operand fragment (32x16 bf16) from B^T stored row-major:
// lane n holds contiguous K 0..15 (lanes 0-15) / 16..31 (lanes 16-31) of col n.
__device__ __forceinline__ v16bf load_b_frag(const bf16* btrow, int hi) {
  return *(const v16bf*)(btrow + hi * 16);
}

__device__ __forceinline__ v8f wmma_bf16(v16bf a, v16bf b, v8f c) {
  return __builtin_amdgcn_wmma_f32_16x16x32_bf16(false, a, false, b, (short)0, c,
                                                 false, false);
}

__device__ __forceinline__ float half_red_max(float v) {
#pragma unroll
  for (int m = 1; m < 16; m <<= 1) v = fmaxf(v, __shfl_xor(v, m, 32));
  return v;
}
__device__ __forceinline__ float half_red_sum(float v) {
#pragma unroll
  for (int m = 1; m < 16; m <<= 1) v += __shfl_xor(v, m, 32);
  return v;
}

// Async 16B global->LDS copy (CDNA5). Flat addresses of __shared__ objects are
// {SHARED_BASE, lds_offset32}, so the low 32 bits are the LDS byte offset.
__device__ __forceinline__ void async_copy_b128(const bf16* lds_dst,
                                                const bf16* gsrc) {
  unsigned loff = (unsigned)(unsigned long long)lds_dst;
  asm volatile("global_load_async_to_lds_b128 %0, %1, off"
               :: "v"(loff), "v"(gsrc)
               : "memory");
}
#define WAIT_ASYNC(imm) asm volatile("s_wait_asynccnt " #imm ::: "memory")
#define WAIT_DS0()      asm volatile("s_wait_dscnt 0x0" ::: "memory")

// ---------------------------------------------------------------------------
// Kernel 1: pack/convert. x -> bf16; Wq/Wk/Wv -> bf16 transposed [H*Dh, E];
// Wd -> bf16 transposed [out=768, in=768].
// ---------------------------------------------------------------------------
#define NX  (B_ * S_ * E_)         // 3145728
#define NW  (H_ * E_ * DH_)        // 589824
#define NWD (DM_ * DM_)            // 589824
#define NPACK (NX + 3 * NW + NWD)  // 5505024

__global__ __launch_bounds__(256) void pack_kernel(
    const float* __restrict__ x, const float* __restrict__ Wq,
    const float* __restrict__ Wk, const float* __restrict__ Wv,
    const float* __restrict__ Wd, bf16* __restrict__ xb,
    bf16* __restrict__ wqT, bf16* __restrict__ wkT, bf16* __restrict__ wvT,
    bf16* __restrict__ wdT) {
  size_t i = (size_t)blockIdx.x * 256 + threadIdx.x;
  if (i < NX) { xb[i] = f2bf(x[i]); return; }
  i -= NX;
  if (i < NW) {  // wqT[(h*64+d)*768+e] = Wq[(h*768+e)*64+d]
    size_t h = i / (DH_ * E_), rem = i % (DH_ * E_), d = rem / E_, e = rem % E_;
    wqT[i] = f2bf(Wq[(h * E_ + e) * DH_ + d]);
    return;
  }
  i -= NW;
  if (i < NW) {
    size_t h = i / (DH_ * E_), rem = i % (DH_ * E_), d = rem / E_, e = rem % E_;
    wkT[i] = f2bf(Wk[(h * E_ + e) * DH_ + d]);
    return;
  }
  i -= NW;
  if (i < NW) {
    size_t h = i / (DH_ * E_), rem = i % (DH_ * E_), d = rem / E_, e = rem % E_;
    wvT[i] = f2bf(Wv[(h * E_ + e) * DH_ + d]);
    return;
  }
  i -= NW;
  if (i < NWD) {  // wdT[o*768+in] = Wd[in*768+o]
    wdT[i] = f2bf(Wd[(i % DM_) * DM_ + (i / DM_)]);
  }
}

// ---------------------------------------------------------------------------
// Kernel 2: QKV projection. Grid = B*H*(S/64), block = 128 (4 waves).
// Weight B-tiles (3 matrices x 64 rows x 32 K) are staged cooperatively into
// double-buffered LDS via global_load_async_to_lds_b128 and shared by all 4
// waves. Q pre-scaled 1/8. Q,K stored [b,h,S,64]; V stored transposed.
// ---------------------------------------------------------------------------
__global__ __launch_bounds__(128) void qkv_kernel(
    const bf16* __restrict__ xb, const bf16* __restrict__ wqT,
    const bf16* __restrict__ wkT, const bf16* __restrict__ wvT,
    const float* __restrict__ bq, const float* __restrict__ bk,
    const float* __restrict__ bv, bf16* __restrict__ Qb, bf16* __restrict__ Kb,
    bf16* __restrict__ VT) {
  __shared__ __align__(64) bf16 wbuf[2][3 * 64 * 32];  // 2 x 12KB

  const int tid = threadIdx.x;
  const int wave = tid >> 5, lane = tid & 31;
  const int lo = lane & 15, hi = lane >> 4;
  const int mblk = blockIdx.x & 15;
  const int bh = blockIdx.x >> 4;
  const int h = bh % H_, b = bh / H_;
  const int m0 = mblk * 64 + wave * 16;

  v8f accq[4] = {}, acck[4] = {}, accv[4] = {};
  const bf16* xrow = xb + (size_t)(b * S_ + m0 + lo) * E_;

  // Issue the 12KB weight slab for k-chunk e0 into buffer p (6 x 16B/thread).
  auto issue_w = [&](int e0, int p) {
#pragma unroll
    for (int m = 0; m < 3; ++m) {
      const bf16* wsrc = (m == 0) ? wqT : (m == 1) ? wkT : wvT;
#pragma unroll
      for (int half = 0; half < 2; ++half) {
        const int c = tid + 128 * half;       // 0..255
        const int row = c >> 2, part = c & 3; // 64 rows x 4 x 16B
        async_copy_b128(&wbuf[p][(m * 64 + row) * 32 + part * 8],
                        wsrc + (size_t)(h * DH_ + row) * E_ + e0 + part * 8);
      }
    }
  };

  issue_w(0, 0);
  for (int step = 0; step < 24; ++step) {
    const int p = step & 1;
    const int e0 = step * 32;
    if (step < 23) { issue_w(e0 + 32, p ^ 1); WAIT_ASYNC(0x6); }
    else           { WAIT_ASYNC(0x0); }
    __syncthreads();                        // slab `step` visible to all waves

    v16bf a = load_a_frag(xrow + e0, hi);
#pragma unroll
    for (int nt = 0; nt < 4; ++nt) {
      const int r = nt * 16 + lo;
      accq[nt] = wmma_bf16(a, load_b_frag(&wbuf[p][(0 * 64 + r) * 32], hi), accq[nt]);
      acck[nt] = wmma_bf16(a, load_b_frag(&wbuf[p][(1 * 64 + r) * 32], hi), acck[nt]);
      accv[nt] = wmma_bf16(a, load_b_frag(&wbuf[p][(2 * 64 + r) * 32], hi), accv[nt]);
    }
    __syncthreads();                        // done reading buf[p] before reuse
  }

  const size_t base = (size_t)(b * H_ + h) * S_ * DH_;
#pragma unroll
  for (int nt = 0; nt < 4; ++nt) {
    const int col = nt * 16 + lo;
    const float vq = bq[h * DH_ + col], vk = bk[h * DH_ + col],
                vv = bv[h * DH_ + col];
#pragma unroll
    for (int r = 0; r < 8; ++r) {
      const int row = m0 + r + hi * 8;
      Qb[base + (size_t)row * DH_ + col] = f2bf((accq[nt][r] + vq) * 0.125f);
      Kb[base + (size_t)row * DH_ + col] = f2bf(acck[nt][r] + vk);
      VT[base + (size_t)col * S_ + row] = f2bf(accv[nt][r] + vv);
    }
  }
}

// ---------------------------------------------------------------------------
// Kernel 3: attention. Grid = B*H*(S/64), block = 128 (4 waves, 16 rows each).
// K tiles (64 seq rows) are async-staged into double-buffered LDS and shared
// by all 4 waves in both passes. Pass A: online softmax stats. Pass B:
// recompute scores, emit normalized probs, relayout P via LDS, ctx = P*V^T.
// ---------------------------------------------------------------------------
__global__ __launch_bounds__(128) void attn_kernel(
    const bf16* __restrict__ Qb, const bf16* __restrict__ Kb,
    const bf16* __restrict__ VT, float* __restrict__ out_w,
    bf16* __restrict__ ctxb) {
  __shared__ __align__(64) bf16 kbuf[2][64 * DH_];   // 2 x 8KB K slabs
  __shared__ __align__(64) bf16 pbuf[4][16 * 32];    // per-wave P relayout

  const int tid = threadIdx.x;
  const int wave = tid >> 5, lane = tid & 31;
  const int lo = lane & 15, hi = lane >> 4;
  const int mblk = blockIdx.x & 15;
  const int bh = blockIdx.x >> 4;
  const int h = bh % H_, b = bh / H_;
  const int m0 = mblk * 64 + wave * 16;
  const size_t base = (size_t)(b * H_ + h) * S_ * DH_;
  const bf16* Kbase = Kb + base;

  const bf16* qrow = Qb + base + (size_t)(m0 + lo) * DH_;
  const v16bf qa0 = load_a_frag(qrow, hi);
  const v16bf qa1 = load_a_frag(qrow + 32, hi);

  // 8KB K slab jb (seq rows [64*jb, 64*jb+64)) into buffer p, 4 x 16B/thread.
  auto issue_k = [&](int jb, int p) {
#pragma unroll
    for (int q = 0; q < 4; ++q) {
      const int c = tid + 128 * q;          // 0..511 x 16B
      async_copy_b128(&kbuf[p][c * 8], Kbase + jb * 64 * DH_ + c * 8);
    }
  };

  float mrun[8], lrun[8];
#pragma unroll
  for (int r = 0; r < 8; ++r) { mrun[r] = -3.0e30f; lrun[r] = 0.0f; }

  // ---- Pass A: softmax statistics -------------------------------------
  issue_k(0, 0);
  for (int jb = 0; jb < 16; ++jb) {
    const int p = jb & 1;
    if (jb < 15) { issue_k(jb + 1, p ^ 1); WAIT_ASYNC(0x4); }
    else         { WAIT_ASYNC(0x0); }
    __syncthreads();
#pragma unroll
    for (int jl = 0; jl < 4; ++jl) {
      const bf16* krow = &kbuf[p][(jl * 16 + lo) * DH_];
      v8f c = {};
      c = wmma_bf16(qa0, load_b_frag(krow, hi), c);
      c = wmma_bf16(qa1, load_b_frag(krow + 32, hi), c);
#pragma unroll
      for (int r = 0; r < 8; ++r) {
        float tmax = half_red_max(c[r]);
        float mnew = fmaxf(mrun[r], tmax);
        float e = __expf(c[r] - mnew);
        float tsum = half_red_sum(e);
        lrun[r] = lrun[r] * __expf(mrun[r] - mnew) + tsum;
        mrun[r] = mnew;
      }
    }
    __syncthreads();
  }
  float linv[8];
#pragma unroll
  for (int r = 0; r < 8; ++r) linv[r] = 1.0f / lrun[r];

  // ---- Pass B: probs out + ctx = P * V^T ------------------------------
  v8f ctx[4] = {};
  bf16* myp = &pbuf[wave][0];

  issue_k(0, 0);
  for (int jb = 0; jb < 16; ++jb) {
    const int p = jb & 1;
    if (jb < 15) { issue_k(jb + 1, p ^ 1); WAIT_ASYNC(0x4); }
    else         { WAIT_ASYNC(0x0); }
    __syncthreads();
#pragma unroll
    for (int jjl = 0; jjl < 2; ++jjl) {
      const int jj = jb * 2 + jjl;          // 32-wide K chunk index
#pragma unroll
      for (int sub = 0; sub < 2; ++sub) {
        const int jl = jjl * 2 + sub;
        const int t0 = (jb * 4 + jl) * 16;
        const bf16* krow = &kbuf[p][(jl * 16 + lo) * DH_];
        v8f c = {};
        c = wmma_bf16(qa0, load_b_frag(krow, hi), c);
        c = wmma_bf16(qa1, load_b_frag(krow + 32, hi), c);
#pragma unroll
        for (int r = 0; r < 8; ++r) {
          const float pv = __expf(c[r] - mrun[r]) * linv[r];
          const int row = m0 + r + hi * 8;
          out_w[(((size_t)b * S_ + row) * H_ + h) * S_ + t0 + lo] = pv;
          myp[(r + hi * 8) * 32 + sub * 16 + lo] = f2bf(pv);
        }
      }
      WAIT_DS0();                           // P tile visible in LDS (per-wave)
      v16bf pa = load_a_frag(myp + lo * 32, hi);
#pragma unroll
      for (int nt = 0; nt < 4; ++nt) {
        const bf16* vrow = VT + base + (size_t)(nt * 16 + lo) * S_ + jj * 32;
        ctx[nt] = wmma_bf16(pa, load_b_frag(vrow, hi), ctx[nt]);
      }
    }
    __syncthreads();
  }

  // ---- epilogue: ctx -> bf16 [B*S, 768] -------------------------------
#pragma unroll
  for (int nt = 0; nt < 4; ++nt) {
    const int col = h * DH_ + nt * 16 + lo;
#pragma unroll
    for (int r = 0; r < 8; ++r) {
      const int row = b * S_ + m0 + r + hi * 8;
      ctxb[(size_t)row * DM_ + col] = f2bf(ctx[nt][r]);
    }
  }
}

// ---------------------------------------------------------------------------
// Kernel 4: dense + bias + LayerNorm. Grid = 4096/16, block = 256 (8 waves).
// Block covers 16 rows x 768 cols; wave w handles cols [96w, 96w+96).
// ---------------------------------------------------------------------------
__global__ __launch_bounds__(256) void dense_ln_kernel(
    const bf16* __restrict__ ctxb, const bf16* __restrict__ wdT,
    const float* __restrict__ bd, const float* __restrict__ gamma,
    const float* __restrict__ beta, float* __restrict__ out_y) {
  __shared__ float red[8][16][2];

  const int wave = threadIdx.x >> 5, lane = threadIdx.x & 31;
  const int lo = lane & 15, hi = lane >> 4;
  const int r0 = blockIdx.x * 16;
  const int nb = wave * 96;

  v8f acc[6] = {};
  const bf16* arow = ctxb + (size_t)(r0 + lo) * DM_;

  for (int e0 = 0; e0 < DM_; e0 += 32) {
    v16bf a = load_a_frag(arow + e0, hi);
#pragma unroll
    for (int nt = 0; nt < 6; ++nt) {
      const bf16* brow = wdT + (size_t)(nb + nt * 16 + lo) * DM_ + e0;
      acc[nt] = wmma_bf16(a, load_b_frag(brow, hi), acc[nt]);
    }
  }

  float psum[8] = {}, psq[8] = {};
#pragma unroll
  for (int nt = 0; nt < 6; ++nt) {
    const float bias = bd[nb + nt * 16 + lo];
#pragma unroll
    for (int r = 0; r < 8; ++r) {
      const float v = acc[nt][r] + bias;
      acc[nt][r] = v;
      psum[r] += v;
      psq[r] += v * v;
    }
  }
#pragma unroll
  for (int r = 0; r < 8; ++r) {
    psum[r] = half_red_sum(psum[r]);
    psq[r] = half_red_sum(psq[r]);
  }
  if (lo == 0) {
#pragma unroll
    for (int r = 0; r < 8; ++r) {
      red[wave][hi * 8 + r][0] = psum[r];
      red[wave][hi * 8 + r][1] = psq[r];
    }
  }
  __syncthreads();

  float mu[8], rs[8];
#pragma unroll
  for (int r = 0; r < 8; ++r) {
    float s = 0.0f, q = 0.0f;
#pragma unroll
    for (int w = 0; w < 8; ++w) {
      s += red[w][hi * 8 + r][0];
      q += red[w][hi * 8 + r][1];
    }
    const float m = s * (1.0f / DM_);
    mu[r] = m;
    rs[r] = rsqrtf(q * (1.0f / DM_) - m * m + LN_EPS);
  }
#pragma unroll
  for (int nt = 0; nt < 6; ++nt) {
    const int col = nb + nt * 16 + lo;
    const float g = gamma[col], be = beta[col];
#pragma unroll
    for (int r = 0; r < 8; ++r) {
      const int row = r0 + hi * 8 + r;
      out_y[(size_t)row * DM_ + col] = (acc[nt][r] - mu[r]) * rs[r] * g + be;
    }
  }
}

// ---------------------------------------------------------------------------
// Launch. Workspace layout (bf16, all offsets multiples of 256 bytes).
// ---------------------------------------------------------------------------
extern "C" void kernel_launch(void* const* d_in, const int* in_sizes, int n_in,
                              void* d_out, int out_size, void* d_ws,
                              size_t ws_size, hipStream_t stream) {
  (void)in_sizes; (void)n_in; (void)out_size; (void)ws_size;

  const float* x  = (const float*)d_in[0];
  const float* Wq = (const float*)d_in[1];
  const float* bq = (const float*)d_in[2];
  const float* Wk = (const float*)d_in[3];
  const float* bk = (const float*)d_in[4];
  const float* Wv = (const float*)d_in[5];
  const float* bv = (const float*)d_in[6];
  const float* Wd = (const float*)d_in[7];
  const float* bd = (const float*)d_in[8];
  const float* gamma = (const float*)d_in[9];
  const float* beta  = (const float*)d_in[10];

  const size_t SZ_XB  = (size_t)B_ * S_ * E_ * 2;   // 6291456
  const size_t SZ_WT  = (size_t)H_ * E_ * DH_ * 2;  // 1179648
  const size_t SZ_WD  = (size_t)DM_ * DM_ * 2;      // 1179648
  const size_t SZ_QKV = (size_t)B_ * H_ * S_ * DH_ * 2;
  const size_t SZ_CTX = (size_t)B_ * S_ * DM_ * 2;

  char* p = (char*)d_ws;
  bf16* xb   = (bf16*)p;                p += SZ_XB;
  bf16* wqT  = (bf16*)p;                p += SZ_WT;
  bf16* wkT  = (bf16*)p;                p += SZ_WT;
  bf16* wvT  = (bf16*)p;                p += SZ_WT;
  bf16* wdT  = (bf16*)p;                p += SZ_WD;
  bf16* Qb   = (bf16*)p;                p += SZ_QKV;
  bf16* Kb   = (bf16*)p;                p += SZ_QKV;
  bf16* VT   = (bf16*)p;                p += SZ_QKV;
  bf16* ctxb = (bf16*)p;                p += SZ_CTX;

  float* out_w = (float*)d_out;                              // [B,S,H*S]
  float* out_y = out_w + (size_t)B_ * S_ * H_ * S_;          // [B,S,768]

  pack_kernel<<<(NPACK + 255) / 256, 256, 0, stream>>>(x, Wq, Wk, Wv, Wd, xb,
                                                       wqT, wkT, wvT, wdT);
  qkv_kernel<<<B_ * H_ * (S_ / 64), 128, 0, stream>>>(xb, wqT, wkT, wvT, bq, bk,
                                                      bv, Qb, Kb, VT);
  attn_kernel<<<B_ * H_ * (S_ / 64), 128, 0, stream>>>(Qb, Kb, VT, out_w, ctxb);
  dense_ln_kernel<<<(B_ * S_) / 16, 256, 0, stream>>>(ctxb, wdT, bd, gamma,
                                                      beta, out_y);
}
